// GRASPLayer_3212635537836
// MI455X (gfx1250) — matrix operand order repaired and checked by reading, not compiled, not run on hardware
//
#include <hip/hip_runtime.h>
#include <hip/hip_bf16.h>
#include <math.h>

// ---------------------------------------------------------------------------
// GRASPLayer for MI455X (gfx1250, wave32, WMMA).
// N=2048, T=128, D=256, H=256, 3H=768, K=12, MAX_ITER=100
// ---------------------------------------------------------------------------

#define N_SEQ   2048
#define T_SEQ   128
#define D_IN    256
#define H_DIM   256
#define G3H     768
#define K_CLU   12
#define KM_ITER 100

typedef __attribute__((ext_vector_type(16))) _Float16 v16h;
typedef __attribute__((ext_vector_type(4)))  _Float16 v4h;
typedef __attribute__((ext_vector_type(8)))  float    v8f;

// ---------------- device helpers ----------------

__device__ __forceinline__ float sigmoidf_(float x) {
  return 1.0f / (1.0f + expf(-x));
}

// A-fragment (16x32 f16, MxK) per CDNA5 ISA layout:
// lane = m + 16*kh ; a[j]   = A[m][kh*8 + j]       (j=0..7)
//                    a[8+j] = A[m][16 + kh*8 + j]
__device__ __forceinline__ v16h load_a_from_f32(const float* __restrict__ rowp, int kh) {
  const float* s0 = rowp + kh * 8;
  const float* s1 = rowp + 16 + kh * 8;
  v16h a;
#pragma unroll
  for (int j = 0; j < 8; ++j) {
    a[j]     = (_Float16)s0[j];
    a[8 + j] = (_Float16)s1[j];
  }
  return a;
}

__device__ __forceinline__ v16h load_a_from_f16(const _Float16* __restrict__ rowp, int kh) {
  const _Float16* s0 = rowp + kh * 8;
  const _Float16* s1 = rowp + 16 + kh * 8;
  v16h a;
#pragma unroll
  for (int j = 0; j < 8; ++j) {
    a[j]     = s0[j];
    a[8 + j] = s1[j];
  }
  return a;
}

// ---------------- weight pack: [768,256] f32 -> B-fragment-ordered f16 -------
// B-fragment (32x16 f16, KxN): lane = n + 16*kh ; b[j] = B[kh*8 + j][n], b[8+j] = B[16 + kh*8 + j][n]
// where B[k][n] = W[jb*16 + n][kc*32 + k].
// Packed layout: tile (jb*8 + kc) is 512 halves, lane-major (32 bytes/lane).
__global__ __launch_bounds__(256) void pack_w_kernel(const float* __restrict__ w,
                                                     _Float16* __restrict__ p) {
  int idx = blockIdx.x * 256 + threadIdx.x;
  if (idx >= G3H * 256) return;
  int j    = idx & 15;
  int lane = (idx >> 4) & 31;
  int tile = idx >> 9;       // 0..383
  int kc   = tile & 7;
  int jb   = tile >> 3;
  int n    = lane & 15;
  int kh   = lane >> 4;
  int wrow = jb * 16 + n;               // output column -> W row
  int wcol = kc * 32 + (j >> 3) * 16 + kh * 8 + (j & 7);     // K index
  p[idx] = (_Float16)w[wrow * 256 + wcol];
}

// ---------------- init h = 0 ----------------
__global__ __launch_bounds__(256) void init_h_kernel(float* __restrict__ h32,
                                                     _Float16* __restrict__ h16) {
  int idx = blockIdx.x * 256 + threadIdx.x;
  if (idx >= N_SEQ * H_DIM) return;
  h32[idx] = 0.0f;
  h16[idx] = (_Float16)0.0f;
}

// ---------------- per-step GEMM: gx = x_t @ W_ih^T ; gh = h @ W_hh^T ---------
// blockIdx.z == 0 : A = x[:, t, :] (fp32 -> f16), B = wih_p, out = gx
// blockIdx.z == 1 : A = h16,                     B = whh_p, out = gh
// 256 threads = 8 waves; each wave computes a 16x64 output strip.
// grid.x = 192 -> 1536 waves = 128 rowGroups x 12 colGroups.
// B tiles double-buffered in registers (software pipeline).

#define WMMA_F16(acc, a, b) \
  __builtin_amdgcn_wmma_f32_16x16x32_f16(false, (a), false, (b), (short)0, (acc), false, false)

__device__ __forceinline__ void gemm_store(
    float* __restrict__ outp, int rowGroup, int jb0, int kh, int m,
    const v8f& acc0, const v8f& acc1, const v8f& acc2, const v8f& acc3) {
  // C/D layout: c[i] = C[kh*8 + i][n], n = lane&15
#pragma unroll
  for (int i = 0; i < 8; ++i) {
    int r = rowGroup * 16 + kh * 8 + i;
    float* orow = outp + (size_t)r * G3H;
    orow[(jb0 + 0) * 16 + m] = acc0[i];
    orow[(jb0 + 1) * 16 + m] = acc1[i];
    orow[(jb0 + 2) * 16 + m] = acc2[i];
    orow[(jb0 + 3) * 16 + m] = acc3[i];
  }
}

__global__ __launch_bounds__(256) void gru_gemm_kernel(
    const float* __restrict__ x, const _Float16* __restrict__ h16,
    const _Float16* __restrict__ wih_p, const _Float16* __restrict__ whh_p,
    float* __restrict__ gx, float* __restrict__ gh, int t) {
  const int lane = threadIdx.x & 31;
  const int gw   = blockIdx.x * 8 + (threadIdx.x >> 5);
  const int rowGroup = gw / 12;   // 0..127
  const int colGroup = gw % 12;   // 0..11
  const int m  = lane & 15;
  const int kh = lane >> 4;
  const int row = rowGroup * 16 + m;
  const int jb0 = colGroup * 4;

  v8f acc0 = {}, acc1 = {}, acc2 = {}, acc3 = {};

  if (blockIdx.z == 0) {
    // ---- A = x[:, t, :] (fp32 -> f16), B = W_ih packed ----
    const float* xrow = x + ((size_t)row * T_SEQ + t) * D_IN;
    const _Float16* bbase = wih_p + (size_t)jb0 * 8 * 512 + lane * 16;
    // tile (u, kc) at bbase + u*4096 + kc*512 (halves)
    v16h b0 = *(const v16h*)(bbase);
    v16h b1 = *(const v16h*)(bbase + 4096);
    v16h b2 = *(const v16h*)(bbase + 8192);
    v16h b3 = *(const v16h*)(bbase + 12288);
#pragma unroll
    for (int kc = 0; kc < 8; ++kc) {
      v16h a = load_a_from_f32(xrow + kc * 32, kh);
      v16h n0 = b0, n1 = b1, n2 = b2, n3 = b3;
      if (kc < 7) {
        const _Float16* nb = bbase + (kc + 1) * 512;
        n0 = *(const v16h*)(nb);
        n1 = *(const v16h*)(nb + 4096);
        n2 = *(const v16h*)(nb + 8192);
        n3 = *(const v16h*)(nb + 12288);
      }
      if (kc < 6) __builtin_prefetch(bbase + (kc + 2) * 512, 0, 3);
      acc0 = WMMA_F16(acc0, a, b0);
      acc1 = WMMA_F16(acc1, a, b1);
      acc2 = WMMA_F16(acc2, a, b2);
      acc3 = WMMA_F16(acc3, a, b3);
      b0 = n0; b1 = n1; b2 = n2; b3 = n3;
    }
    gemm_store(gx, rowGroup, jb0, kh, m, acc0, acc1, acc2, acc3);
  } else {
    // ---- A = h (f16), B = W_hh packed ----
    const _Float16* hrow = h16 + (size_t)row * H_DIM;
    const _Float16* bbase = whh_p + (size_t)jb0 * 8 * 512 + lane * 16;
    v16h b0 = *(const v16h*)(bbase);
    v16h b1 = *(const v16h*)(bbase + 4096);
    v16h b2 = *(const v16h*)(bbase + 8192);
    v16h b3 = *(const v16h*)(bbase + 12288);
#pragma unroll
    for (int kc = 0; kc < 8; ++kc) {
      v16h a = load_a_from_f16(hrow + kc * 32, kh);
      v16h n0 = b0, n1 = b1, n2 = b2, n3 = b3;
      if (kc < 7) {
        const _Float16* nb = bbase + (kc + 1) * 512;
        n0 = *(const v16h*)(nb);
        n1 = *(const v16h*)(nb + 4096);
        n2 = *(const v16h*)(nb + 8192);
        n3 = *(const v16h*)(nb + 12288);
      }
      if (kc < 6) __builtin_prefetch(bbase + (kc + 2) * 512, 0, 3);
      acc0 = WMMA_F16(acc0, a, b0);
      acc1 = WMMA_F16(acc1, a, b1);
      acc2 = WMMA_F16(acc2, a, b2);
      acc3 = WMMA_F16(acc3, a, b3);
      b0 = n0; b1 = n1; b2 = n2; b3 = n3;
    }
    gemm_store(gh, rowGroup, jb0, kh, m, acc0, acc1, acc2, acc3);
  }
}

// ---------------- GRU gate combine (elementwise, float4-vectorized) ----------
__global__ __launch_bounds__(256) void gru_combine_kernel(
    const float* __restrict__ gx, const float* __restrict__ gh,
    const float* __restrict__ bih, const float* __restrict__ bhh,
    const int* __restrict__ lengths,
    float* __restrict__ h32, _Float16* __restrict__ h16,
    float* __restrict__ hfin, int t) {
  int idx = blockIdx.x * 256 + threadIdx.x;   // < N*H/4
  if (idx >= N_SEQ * H_DIM / 4) return;
  int i = idx >> 6;
  int j = (idx & 63) * 4;
  size_t g = (size_t)i * G3H + j;
  size_t hidx = (size_t)i * H_DIM + j;

  float4 xr = *(const float4*)(gx + g);
  float4 xz = *(const float4*)(gx + g + 256);
  float4 xn = *(const float4*)(gx + g + 512);
  float4 hr = *(const float4*)(gh + g);
  float4 hz = *(const float4*)(gh + g + 256);
  float4 hn = *(const float4*)(gh + g + 512);
  float4 bir = *(const float4*)(bih + j);
  float4 biz = *(const float4*)(bih + j + 256);
  float4 bin = *(const float4*)(bih + j + 512);
  float4 bhr = *(const float4*)(bhh + j);
  float4 bhz = *(const float4*)(bhh + j + 256);
  float4 bhn = *(const float4*)(bhh + j + 512);
  float4 hp = *(const float4*)(h32 + hidx);

  float4 hnew;
  v4h hnew16;
  {
    float r = sigmoidf_((xr.x + bir.x) + (hr.x + bhr.x));
    float z = sigmoidf_((xz.x + biz.x) + (hz.x + bhz.x));
    float n = tanhf((xn.x + bin.x) + r * (hn.x + bhn.x));
    hnew.x = (1.0f - z) * n + z * hp.x;
  }
  {
    float r = sigmoidf_((xr.y + bir.y) + (hr.y + bhr.y));
    float z = sigmoidf_((xz.y + biz.y) + (hz.y + bhz.y));
    float n = tanhf((xn.y + bin.y) + r * (hn.y + bhn.y));
    hnew.y = (1.0f - z) * n + z * hp.y;
  }
  {
    float r = sigmoidf_((xr.z + bir.z) + (hr.z + bhr.z));
    float z = sigmoidf_((xz.z + biz.z) + (hz.z + bhz.z));
    float n = tanhf((xn.z + bin.z) + r * (hn.z + bhn.z));
    hnew.z = (1.0f - z) * n + z * hp.z;
  }
  {
    float r = sigmoidf_((xr.w + bir.w) + (hr.w + bhr.w));
    float z = sigmoidf_((xz.w + biz.w) + (hz.w + bhz.w));
    float n = tanhf((xn.w + bin.w) + r * (hn.w + bhn.w));
    hnew.w = (1.0f - z) * n + z * hp.w;
  }
  hnew16[0] = (_Float16)hnew.x;
  hnew16[1] = (_Float16)hnew.y;
  hnew16[2] = (_Float16)hnew.z;
  hnew16[3] = (_Float16)hnew.w;

  *(float4*)(h32 + hidx) = hnew;
  *(v4h*)(h16 + hidx) = hnew16;
  if (t + 1 == lengths[i]) *(float4*)(hfin + hidx) = hnew;
}

// ---------------- k-means (multi-block per iteration) ----------------
__global__ __launch_bounds__(256) void kmeans_init_kernel(
    const float* __restrict__ data, float* __restrict__ centers,
    float* __restrict__ gsums, float* __restrict__ gcnts) {
  int i = blockIdx.x * 256 + threadIdx.x;
  if (i >= K_CLU * H_DIM) return;
  int c = i >> 8, d = i & 255;
  int src = (c * 151 + 7) & (N_SEQ - 1);
  centers[i] = data[(size_t)src * H_DIM + d];
  gsums[i] = 0.0f;
  if (i < K_CLU) gcnts[i] = 0.0f;
}

// assign + partial segment-sums. One wave32 per row; lane owns 8 dims.
__global__ __launch_bounds__(256) void kmeans_assign_kernel(
    const float* __restrict__ data, const float* __restrict__ centers,
    float* __restrict__ gsums, float* __restrict__ gcnts) {
  __shared__ float lsum[K_CLU * H_DIM];
  __shared__ float lcnt[K_CLU];
  const int tid  = threadIdx.x;
  const int lane = tid & 31;
  const int wv   = tid >> 5;

  for (int i = tid; i < K_CLU * H_DIM; i += 256) lsum[i] = 0.0f;
  if (tid < K_CLU) lcnt[tid] = 0.0f;
  __syncthreads();

  float cr[K_CLU][8];
#pragma unroll
  for (int j = 0; j < K_CLU; ++j)
#pragma unroll
    for (int q = 0; q < 8; ++q) cr[j][q] = centers[j * H_DIM + lane + 32 * q];

  for (int rr = 0; rr < 8; ++rr) {
    int row = blockIdx.x * 64 + rr * 8 + wv;
    const float* hrow = data + (size_t)row * H_DIM;
    float hv[8];
#pragma unroll
    for (int q = 0; q < 8; ++q) hv[q] = hrow[lane + 32 * q];

    float best = 3.4e38f;
    int bj = 0;
#pragma unroll
    for (int j = 0; j < K_CLU; ++j) {
      float p = 0.0f;
#pragma unroll
      for (int q = 0; q < 8; ++q) {
        float df = hv[q] - cr[j][q];
        p = fmaf(df, df, p);
      }
      for (int s = 16; s > 0; s >>= 1) p += __shfl_xor(p, s, 32);
      if (p < best) { best = p; bj = j; }   // strict < => lowest index on ties
    }
#pragma unroll
    for (int q = 0; q < 8; ++q) atomicAdd(&lsum[bj * H_DIM + lane + 32 * q], hv[q]);
    if (lane == 0) atomicAdd(&lcnt[bj], 1.0f);
  }
  __syncthreads();

  for (int i = tid; i < K_CLU * H_DIM; i += 256) atomicAdd(&gsums[i], lsum[i]);
  if (tid < K_CLU) atomicAdd(&gcnts[tid], lcnt[tid]);
}

// update: block b owns cluster b (256 dims); re-zeros its accumulators.
__global__ __launch_bounds__(256) void kmeans_update_kernel(
    float* __restrict__ centers, float* __restrict__ gsums,
    float* __restrict__ gcnts) {
  const int c = blockIdx.x;
  const int d = threadIdx.x;
  const int i = c * H_DIM + d;
  float cnt = gcnts[c];
  float s = gsums[i];
  if (cnt > 0.0f) centers[i] = s / cnt;
  gsums[i] = 0.0f;
  __syncthreads();
  if (d == 0) gcnts[c] = 0.0f;
}

// ---------------- tiny 2-layer GCN (adj = I) ----------------
__global__ __launch_bounds__(1024) void gcn_kernel(
    const float* __restrict__ centers,
    const float* __restrict__ g1w, const float* __restrict__ g1b,
    const float* __restrict__ g2w, const float* __restrict__ g2b,
    float* __restrict__ h2out) {
  __shared__ float h1[K_CLU * H_DIM];
  const int tid = threadIdx.x;
  for (int i = tid; i < K_CLU * H_DIM; i += 1024) {
    int c = i >> 8, o = i & 255;
    float acc = g1b[o];
    for (int d = 0; d < H_DIM; ++d) acc = fmaf(centers[c * H_DIM + d], g1w[d * H_DIM + o], acc);
    h1[i] = fmaxf(acc, 0.0f);
  }
  __syncthreads();
  for (int i = tid; i < K_CLU * H_DIM; i += 1024) {
    int c = i >> 8, o = i & 255;
    float acc = g2b[o];
    for (int d = 0; d < H_DIM; ++d) acc = fmaf(h1[c * H_DIM + d], g2w[d * H_DIM + o], acc);
    h2out[i] = fmaxf(acc, 0.0f);
  }
}

// ---------------- final mix: softmax attention over clusters + gating --------
__global__ __launch_bounds__(256) void final_kernel(
    const float* __restrict__ hfin, const float* __restrict__ centers,
    const float* __restrict__ h2,
    const float* __restrict__ w1w, const float* __restrict__ w1b,
    const float* __restrict__ w2w, const float* __restrict__ w2b,
    float* __restrict__ out) {
  const int lane = threadIdx.x & 31;
  const int row  = blockIdx.x * 8 + (threadIdx.x >> 5);
  const float* h = hfin + (size_t)row * H_DIM;

  float hv[8];
#pragma unroll
  for (int q = 0; q < 8; ++q) hv[q] = h[lane + 32 * q];

  float e[K_CLU];
#pragma unroll
  for (int j = 0; j < K_CLU; ++j) {
    float p = 0.0f;
#pragma unroll
    for (int q = 0; q < 8; ++q) p = fmaf(hv[q], centers[j * H_DIM + lane + 32 * q], p);
    for (int s = 16; s > 0; s >>= 1) p += __shfl_xor(p, s, 32);
    e[j] = fmaxf(p, 0.0f);  // relu
  }
  float mx = e[0];
#pragma unroll
  for (int j = 1; j < K_CLU; ++j) mx = fmaxf(mx, e[j]);
  float sum = 0.0f;
#pragma unroll
  for (int j = 0; j < K_CLU; ++j) { e[j] = expf(e[j] - mx); sum += e[j]; }
  float inv = 1.0f / sum;

  float ca[8];
#pragma unroll
  for (int q = 0; q < 8; ++q) ca[q] = 0.0f;
#pragma unroll
  for (int j = 0; j < K_CLU; ++j) {
    float s = e[j] * inv;
#pragma unroll
    for (int q = 0; q < 8; ++q) ca[q] = fmaf(s, h2[j * H_DIM + lane + 32 * q], ca[q]);
  }

  float p1 = 0.0f, p2 = 0.0f;
#pragma unroll
  for (int q = 0; q < 8; ++q) {
    int d = lane + 32 * q;
    p1 = fmaf(ca[q], w1w[d], p1);
    p2 = fmaf(hv[q], w2w[d], p2);
  }
  for (int s = 16; s > 0; s >>= 1) { p1 += __shfl_xor(p1, s, 32); p2 += __shfl_xor(p2, s, 32); }

  float w1 = sigmoidf_(p1 + w1b[0]);
  float w2 = sigmoidf_(p2 + w2b[0]);
  float w1n = w1 / (w1 + w2);

  float* orow = out + (size_t)row * H_DIM;
#pragma unroll
  for (int q = 0; q < 8; ++q)
    orow[lane + 32 * q] = w1n * ca[q] + (1.0f - w1n) * hv[q];
}

// ---------------------------------------------------------------------------
extern "C" void kernel_launch(void* const* d_in, const int* in_sizes, int n_in,
                              void* d_out, int out_size, void* d_ws, size_t ws_size,
                              hipStream_t stream) {
  const float* x       = (const float*)d_in[0];
  const int*   lengths = (const int*)  d_in[1];
  const float* W_ih    = (const float*)d_in[2];
  const float* W_hh    = (const float*)d_in[3];
  const float* b_ih    = (const float*)d_in[4];
  const float* b_hh    = (const float*)d_in[5];
  const float* w1_w    = (const float*)d_in[6];
  const float* w1_b    = (const float*)d_in[7];
  const float* w2_w    = (const float*)d_in[8];
  const float* w2_b    = (const float*)d_in[9];
  const float* g1_w    = (const float*)d_in[10];
  const float* g1_b    = (const float*)d_in[11];
  const float* g2_w    = (const float*)d_in[12];
  const float* g2_b    = (const float*)d_in[13];
  float* out = (float*)d_out;

  char* ws = (char*)d_ws;
  size_t off = 0;
  auto carve = [&](size_t bytes) -> char* {
    char* p = ws + off;
    off = (off + bytes + 255) & ~(size_t)255;
    return p;
  };
  _Float16* wih_p   = (_Float16*)carve((size_t)G3H * 256 * 2);
  _Float16* whh_p   = (_Float16*)carve((size_t)G3H * 256 * 2);
  _Float16* h16     = (_Float16*)carve((size_t)N_SEQ * H_DIM * 2);
  float*    h32     = (float*)   carve((size_t)N_SEQ * H_DIM * 4);
  float*    hfin    = (float*)   carve((size_t)N_SEQ * H_DIM * 4);
  float*    gx      = (float*)   carve((size_t)N_SEQ * G3H * 4);
  float*    gh      = (float*)   carve((size_t)N_SEQ * G3H * 4);
  float*    centers = (float*)   carve((size_t)K_CLU * H_DIM * 4);
  float*    h2mat   = (float*)   carve((size_t)K_CLU * H_DIM * 4);
  float*    gsums   = (float*)   carve((size_t)K_CLU * H_DIM * 4);
  float*    gcnts   = (float*)   carve((size_t)K_CLU * 4);

  // 1) pack weights to WMMA B-fragment order (f16)
  pack_w_kernel<<<G3H * 256 / 256, 256, 0, stream>>>(W_ih, wih_p);
  pack_w_kernel<<<G3H * 256 / 256, 256, 0, stream>>>(W_hh, whh_p);

  // 2) h0 = 0
  init_h_kernel<<<N_SEQ * H_DIM / 256, 256, 0, stream>>>(h32, h16);

  // 3) GRU scan (sequential in t; each step is WMMA GEMMs + gate fuse)
  for (int t = 0; t < T_SEQ; ++t) {
    gru_gemm_kernel<<<dim3(192, 1, 2), 256, 0, stream>>>(x, h16, wih_p, whh_p, gx, gh, t);
    gru_combine_kernel<<<N_SEQ * H_DIM / 4 / 256, 256, 0, stream>>>(
        gx, gh, b_ih, b_hh, lengths, h32, h16, hfin, t);
  }

  // 4) k-means on final hidden states (100 iterations, chip-wide per iter)
  kmeans_init_kernel<<<(K_CLU * H_DIM + 255) / 256, 256, 0, stream>>>(hfin, centers, gsums, gcnts);
  for (int it = 0; it < KM_ITER; ++it) {
    kmeans_assign_kernel<<<N_SEQ / 64, 256, 0, stream>>>(hfin, centers, gsums, gcnts);
    kmeans_update_kernel<<<K_CLU, 256, 0, stream>>>(centers, gsums, gcnts);
  }

  // 5) tiny GCN on centers
  gcn_kernel<<<1, 1024, 0, stream>>>(centers, g1_w, g1_b, g2_w, g2_b, h2mat);

  // 6) softmax attention + gated mix -> output [N, H]
  final_kernel<<<N_SEQ / 8, 256, 0, stream>>>(hfin, centers, h2mat,
                                              w1_w, w1_b, w2_w, w2_b, out);
}
